// DetailTransformer_41223096107636
// MI455X (gfx1250) — matrix operand order repaired and checked by reading, not compiled, not run on hardware
//
#include <hip/hip_runtime.h>
#include <math.h>
#include <stdint.h>

// ---------------------------------------------------------------------------
// Music-Transformer forward for MI455X (gfx1250, wave32, WMMA).
// GEMMs: v_wmma_f32_16x16x32_f16, 4x2 tiles per wave, b128 fragment loads,
// pre-packed B-fragment weights, global_prefetch. Attention stages Q via
// global_load_async_to_lds_b128 (ASYNCcnt) and fuses QK^T + skewed rel-pos
// GEMM + softmax + PV entirely in LDS.
// ---------------------------------------------------------------------------

typedef __attribute__((ext_vector_type(16))) _Float16 v16h;
typedef __attribute__((ext_vector_type(8)))  _Float16 v8h;
typedef __attribute__((ext_vector_type(8)))  float    v8f;

#define NB      8
#define TSEQ    512
#define BT      4096          // NB*TSEQ
#define NH      8
#define DMODEL  512
#define HD      64
#define DFFN    2048
#define NLAYER  6
#define DOF     69

static __device__ __forceinline__ int imin(int a, int b) { return a < b ? a : b; }

// -------------------------------- helpers ----------------------------------

// plain row-major f32 -> f16 with zero padding (activations)
__global__ void cvt_pad_f16(const float* __restrict__ src, _Float16* __restrict__ dst,
                            int Kin, int Nin, int Kpad, int Npad) {
  int idx = blockIdx.x * 256 + threadIdx.x;
  int total = Kpad * Npad;
  if (idx >= total) return;
  int r = idx / Npad, c = idx % Npad;
  dst[idx] = (r < Kin && c < Nin) ? (_Float16)src[(size_t)r * Nin + c] : (_Float16)0.0f;
}

// weight pack: W(K x N) f32 -> Wp[(k/32)*Npad + n][k%32] f16 (B-fragment layout)
__global__ void cvt_pack_b(const float* __restrict__ src, _Float16* __restrict__ dst,
                           int Kin, int Nin, int Kpad, int Npad) {
  int idx = blockIdx.x * 256 + threadIdx.x;
  int total = Kpad * Npad;
  if (idx >= total) return;
  int k = idx / Npad, n = idx % Npad;
  float v = (k < Kin && n < Nin) ? src[(size_t)k * Nin + n] : 0.f;
  dst[(((size_t)(k >> 5)) * Npad + n) * 32 + (k & 31)] = (_Float16)v;
}

__global__ void pack_enc(const float* __restrict__ x, const float* __restrict__ mask,
                         _Float16* __restrict__ dst) {
  int idx = blockIdx.x * 256 + threadIdx.x;
  if (idx >= BT * 160) return;
  int bt = idx / 160, c = idx % 160;
  float v = 0.f;
  if (c < DOF)            v = x[(size_t)bt * DOF + c];
  else if (c < 2 * DOF)   v = mask[(size_t)bt * DOF + (c - DOF)];
  dst[idx] = (_Float16)v;
}

// hidden[r,n] = prelu((r-511)*rp_w1[n] + rp_b1[n], *rp_a), r in [0,1023)
__global__ void rp_hidden(const float* __restrict__ w1, const float* __restrict__ b1,
                          const float* __restrict__ a, _Float16* __restrict__ dst) {
  int idx = blockIdx.x * 256 + threadIdx.x;
  if (idx >= 1023 * DMODEL) return;
  int r = idx / DMODEL, n = idx % DMODEL;
  float rel = (float)(r - 511);
  float v = rel * w1[n] + b1[n];
  float al = a[0];
  v = (v >= 0.f) ? v : al * v;
  dst[idx] = (_Float16)v;
}

// one row (512) per block, 256 threads, f16 normalized output
__global__ void layernorm_f16(const float* __restrict__ h, const float* __restrict__ gamma,
                              const float* __restrict__ beta, _Float16* __restrict__ out) {
  __shared__ float red[256];
  int row = blockIdx.x, tid = threadIdx.x;
  const float* xr = h + (size_t)row * DMODEL;
  float v0 = xr[tid], v1 = xr[tid + 256];
  red[tid] = v0 + v1;
  __syncthreads();
  for (int st = 128; st > 0; st >>= 1) { if (tid < st) red[tid] += red[tid + st]; __syncthreads(); }
  float mu = red[0] * (1.f / 512.f);
  __syncthreads();
  float d0 = v0 - mu, d1 = v1 - mu;
  red[tid] = d0 * d0 + d1 * d1;
  __syncthreads();
  for (int st = 128; st > 0; st >>= 1) { if (tid < st) red[tid] += red[tid + st]; __syncthreads(); }
  float inv = rsqrtf(red[0] * (1.f / 512.f) + 1e-6f);
  out[(size_t)row * DMODEL + tid]       = (_Float16)(d0 * inv * gamma[tid]       + beta[tid]);
  out[(size_t)row * DMODEL + tid + 256] = (_Float16)(d1 * inv * gamma[tid + 256] + beta[tid + 256]);
}

// ------------------------------ generic GEMM --------------------------------
// C(MxN) = A(MxK,f16 row-major) @ Wp(packed B-frag layout, Npack cols)
// [+bias] [prelu(alpha)] [+resid]; outputs: f32 / f16 / f16-transposed (V) /
// final (split 69 raw + 4 sigmoid into d_out).
// Block 128 threads = 4 waves; block tile 128x64; each wave owns 4x2 WMMA
// tiles -> 8 wmma per k-step against 6 b128-pair fragment loads.
__global__ void gemm_wmma_kernel(const _Float16* __restrict__ A,
                                 const _Float16* __restrict__ Wp,
                                 const float* __restrict__ bias,
                                 const float* __restrict__ alpha,
                                 const float* __restrict__ resid,
                                 float* __restrict__ outF,
                                 _Float16* __restrict__ outH,
                                 _Float16* __restrict__ outHT,
                                 float* __restrict__ finalO,
                                 int M, int N, int K, int Npack) {
  const int tid  = threadIdx.x;
  const int lane = tid & 31;
  const int wave = tid >> 5;
  const int ln16 = lane & 15;
  const int g    = (lane >> 4) & 1;
  const int kh   = g * 8;

  const int mBase = blockIdx.y * 128 + (wave >> 1) * 64;
  const int nBase = blockIdx.x * 64  + (wave & 1) * 32;

  // clamped (branch-free) load indices: out-of-range rows/cols only feed
  // store-guarded D rows/cols, so reading valid-but-wrong data is harmless.
  const _Float16* aRow[4];
  size_t bCol[2];
#pragma unroll
  for (int rt = 0; rt < 4; rt++)
    aRow[rt] = A + (size_t)imin(mBase + rt * 16 + ln16, M - 1) * K;
#pragma unroll
  for (int ct = 0; ct < 2; ct++)
    bCol[ct] = (size_t)imin(nBase + ct * 16 + ln16, Npack - 1) * 32 + g * 16;

  v8f acc[4][2];
#pragma unroll
  for (int rt = 0; rt < 4; rt++)
#pragma unroll
    for (int ct = 0; ct < 2; ct++)
      for (int i = 0; i < 8; i++) acc[rt][ct][i] = 0.f;

  for (int k0 = 0; k0 < K; k0 += 32) {
    const size_t panel = ((size_t)(k0 >> 5)) * Npack * 32;
    v16h afr[4], bfr[2];
#pragma unroll
    for (int rt = 0; rt < 4; rt++) {
      v8h lo = *(const v8h*)(aRow[rt] + k0 + g * 16);        // K = 16g..16g+7
      v8h hi = *(const v8h*)(aRow[rt] + k0 + g * 16 + 8);    // K = 16g+8..16g+15
#pragma unroll
      for (int i = 0; i < 8; i++) { afr[rt][i] = lo[i]; afr[rt][8 + i] = hi[i]; }
    }
#pragma unroll
    for (int ct = 0; ct < 2; ct++)
      bfr[ct] = *(const v16h*)(Wp + panel + bCol[ct]);   // 16 contiguous halves

    __builtin_prefetch((const void*)(Wp + panel + (size_t)Npack * 32), 0, 1);

#pragma unroll
    for (int rt = 0; rt < 4; rt++)
#pragma unroll
      for (int ct = 0; ct < 2; ct++)
        acc[rt][ct] = __builtin_amdgcn_wmma_f32_16x16x32_f16(
            false, afr[rt], false, bfr[ct], (short)0, acc[rt][ct], false, false);
  }

  float al = (alpha != nullptr) ? alpha[0] : 0.f;
  for (int rt = 0; rt < 4; rt++)
    for (int ct = 0; ct < 2; ct++) {
      int n = nBase + ct * 16 + ln16;
      if (n >= N) continue;
      for (int i = 0; i < 8; i++) {
        int m = mBase + rt * 16 + i + 8 * g;
        if (m >= M) continue;
        float v = acc[rt][ct][i];
        if (bias)  v += bias[n];
        if (alpha) v = (v >= 0.f) ? v : al * v;
        if (resid) v += resid[(size_t)m * N + n];
        if (outH)  outH[(size_t)m * N + n] = (_Float16)v;
        if (outHT) // V: [b*NH + head][headdim][seq]
          outHT[(((size_t)(m >> 9) * NH + (n >> 6)) * HD + (n & 63)) * TSEQ + (m & 511)] =
              (_Float16)v;
        if (outF)  outF[(size_t)m * N + n] = v;
        if (finalO) {
          if (n < DOF) finalO[(size_t)m * DOF + n] = v;
          else         finalO[(size_t)BT * DOF + (size_t)m * 4 + (n - DOF)] =
                         1.f / (1.f + expf(-v));
        }
      }
    }
}

// ------------------------------ fused attention -----------------------------
// One block per (b, h, 64-query tile). S = Q K^T (WMMA) + skewed Q E^T (WMMA),
// row softmax, O = P V (WMMA). Dynamic LDS: S 64x520 f32, G 16x544 f32, Q 64x64 f16.
#define S_LD   520
#define G_LD   544
#define ATT_SMEM_BYTES (64 * S_LD * 4 + 16 * G_LD * 4 + 64 * 64 * 2)

__global__ void attention_wmma(const _Float16* __restrict__ q16,
                               const _Float16* __restrict__ k16,
                               const _Float16* __restrict__ vT,   // [bh][hd][seq]
                               const _Float16* __restrict__ E16,  // [1023][64]
                               _Float16* __restrict__ o16) {
  const int bh = blockIdx.y;       // 0..63
  const int b  = bh >> 3;
  const int hh = bh & 7;
  const int t0 = blockIdx.x * 64;

  const int tid  = threadIdx.x;    // 256 threads = 8 waves
  const int lane = tid & 31;
  const int wave = tid >> 5;
  const int ln16 = lane & 15;
  const int g    = (lane >> 4) & 1;
  const int kh   = g * 8;

  extern __shared__ char smem[];
  float*    S  = (float*)smem;                                        // 64 x S_LD
  float*    G  = (float*)(smem + 64 * S_LD * 4);                      // 16 x G_LD
  _Float16* Qs = (_Float16*)(smem + 64 * S_LD * 4 + 16 * G_LD * 4);   // 64 x 64

  // ---- stage Q tile: async global->LDS DMA (b128 per lane, ASYNCcnt) ----
  {
    const unsigned long long qbase =
        (unsigned long long)(uintptr_t)(q16 + ((size_t)(b * TSEQ + t0)) * DMODEL + hh * HD);
#pragma unroll
    for (int it = 0; it < 2; ++it) {
      int idx = it * 256 + tid;                 // 0..511 : 512 x 16B = 8KB
      int r = idx >> 3, c8 = (idx & 7) * 8;
      unsigned int goff  = (unsigned int)((r * DMODEL + c8) * 2);   // bytes
      unsigned int laddr = (unsigned int)(uintptr_t)(Qs + r * 64 + c8);
      asm volatile("global_load_async_to_lds_b128 %0, %1, %2 offset:0"
                   :: "v"(laddr), "v"(goff), "s"(qbase) : "memory");
    }
    asm volatile("s_wait_asynccnt 0x0" ::: "memory");
  }
  __syncthreads();

  // ---- S = Q @ K^T : 4 x 32 tiles of 16x16, K-dim 64 (2 wmma each) ----
  for (int ti = wave; ti < 128; ti += 8) {
    int rt = ti >> 5, ct = ti & 31;
    int s  = ct * 16 + ln16;                 // key index (column of S)
    const _Float16* kRow = k16 + ((size_t)(b * TSEQ + s)) * DMODEL + hh * HD;
    v8f acc; for (int i = 0; i < 8; i++) acc[i] = 0.f;
#pragma unroll
    for (int k0 = 0; k0 < HD; k0 += 32) {
      v16h a, bfr;
      v8h lo = *(const v8h*)(Qs + (rt * 16 + ln16) * 64 + k0 + g * 16);
      v8h hi = *(const v8h*)(Qs + (rt * 16 + ln16) * 64 + k0 + g * 16 + 8);
#pragma unroll
      for (int i = 0; i < 8; i++) { a[i] = lo[i]; a[8 + i] = hi[i]; }
      bfr = *(const v16h*)(kRow + k0 + g * 16);  // 16 contiguous features
      acc = __builtin_amdgcn_wmma_f32_16x16x32_f16(false, a, false, bfr, (short)0, acc, false, false);
    }
    for (int i = 0; i < 8; i++)
      S[(rt * 16 + i + 8 * g) * S_LD + ct * 16 + ln16] = acc[i];
  }
  __syncthreads();

  // ---- relative term: per 16-row group, G = Q16 @ E_band^T, diagonal add ----
  for (int sub = 0; sub < 4; ++sub) {
    int t0g  = t0 + sub * 16;
    int r_lo = 496 - t0g;                     // always >= 0
    for (int gt = wave; gt < 34; gt += 8) {   // 34 column tiles -> 544 cols
      // clamp r: columns with r > 1022 are never consumed by the skew add
      int r = imin(r_lo + gt * 16 + ln16, 1022);
      const _Float16* eRow = E16 + (size_t)r * HD;
      v8f acc; for (int i = 0; i < 8; i++) acc[i] = 0.f;
#pragma unroll
      for (int k0 = 0; k0 < HD; k0 += 32) {
        v16h a, bfr;
        v8h lo = *(const v8h*)(Qs + (sub * 16 + ln16) * 64 + k0 + g * 16);
        v8h hi = *(const v8h*)(Qs + (sub * 16 + ln16) * 64 + k0 + g * 16 + 8);
#pragma unroll
        for (int i = 0; i < 8; i++) { a[i] = lo[i]; a[8 + i] = hi[i]; }
        bfr = *(const v16h*)(eRow + k0 + g * 16);
        acc = __builtin_amdgcn_wmma_f32_16x16x32_f16(false, a, false, bfr, (short)0, acc, false, false);
      }
      for (int i = 0; i < 8; i++)
        G[(i + 8 * g) * G_LD + gt * 16 + ln16] = acc[i];
    }
    __syncthreads();
    // S[t0g+i, s] += G[i, s - i + 15]   (skew: r = s - t + 511)
    for (int idx = tid; idx < 16 * TSEQ; idx += 256) {
      int i = idx >> 9, s = idx & 511;
      S[(sub * 16 + i) * S_LD + s] += G[i * G_LD + (s - i + 15)];
    }
    __syncthreads();
  }

  // ---- softmax over 512 cols, scale = 1/sqrt(64) ----
  {
    const float scale = 0.125f;
    int row = tid >> 2, part = tid & 3;       // 4 threads per row
    float* red = G;                           // reuse: 64 rows x 4 floats
    float lmax = -3.0e38f;
    for (int s = part * 128; s < part * 128 + 128; ++s)
      lmax = fmaxf(lmax, S[row * S_LD + s] * scale);
    red[row * 4 + part] = lmax;
    __syncthreads();
    float rmax = fmaxf(fmaxf(red[row * 4 + 0], red[row * 4 + 1]),
                       fmaxf(red[row * 4 + 2], red[row * 4 + 3]));
    __syncthreads();
    float lsum = 0.f;
    for (int s = part * 128; s < part * 128 + 128; ++s) {
      float e = expf(S[row * S_LD + s] * scale - rmax);
      S[row * S_LD + s] = e;
      lsum += e;
    }
    red[row * 4 + part] = lsum;
    __syncthreads();
    float inv = 1.f / (red[row * 4 + 0] + red[row * 4 + 1] +
                       red[row * 4 + 2] + red[row * 4 + 3]);
    for (int s = part * 128; s < part * 128 + 128; ++s)
      S[row * S_LD + s] *= inv;
  }
  __syncthreads();

  // ---- O = P @ V : 4 x 4 tiles, K-dim 512 (16 wmma each) ----
  for (int ti = wave; ti < 16; ti += 8) {
    int rt = ti >> 2, ct = ti & 3;
    int n  = ct * 16 + ln16;                  // head-dim column
    const _Float16* vRow = vT + ((size_t)(bh * HD + n)) * TSEQ;  // contiguous in seq
    v8f acc; for (int i = 0; i < 8; i++) acc[i] = 0.f;
    for (int k0 = 0; k0 < TSEQ; k0 += 32) {
      v16h a, bfr;
      v8f plo = *(const v8f*)(S + (rt * 16 + ln16) * S_LD + k0 + g * 16);
      v8f phi = *(const v8f*)(S + (rt * 16 + ln16) * S_LD + k0 + g * 16 + 8);
#pragma unroll
      for (int i = 0; i < 8; i++) { a[i] = (_Float16)plo[i]; a[8 + i] = (_Float16)phi[i]; }
      bfr = *(const v16h*)(vRow + k0 + g * 16);
      acc = __builtin_amdgcn_wmma_f32_16x16x32_f16(false, a, false, bfr, (short)0, acc, false, false);
    }
    for (int i = 0; i < 8; i++)
      o16[((size_t)(b * TSEQ + t0 + rt * 16 + i + 8 * g)) * DMODEL + hh * HD + n] =
          (_Float16)acc[i];
  }
}

// -------------------------------- launcher ----------------------------------

extern "C" void kernel_launch(void* const* d_in, const int* in_sizes, int n_in,
                              void* d_out, int out_size, void* d_ws, size_t ws_size,
                              hipStream_t stream) {
  (void)in_sizes; (void)n_in; (void)out_size; (void)ws_size;

  const float* x_in    = (const float*)d_in[0];
  const float* mask_in = (const float*)d_in[1];
  const float* enc_w1  = (const float*)d_in[2];
  const float* enc_b1  = (const float*)d_in[3];
  const float* enc_a1  = (const float*)d_in[4];
  const float* enc_w2  = (const float*)d_in[5];
  const float* enc_b2  = (const float*)d_in[6];
  const float* enc_a2  = (const float*)d_in[7];
  const float* rp_w1   = (const float*)d_in[8];
  const float* rp_b1   = (const float*)d_in[9];
  const float* rp_a    = (const float*)d_in[10];
  const float* rp_w2   = (const float*)d_in[11];
  const float* rp_b2   = (const float*)d_in[12];
  const float* ln_g    = (const float*)d_in[13];
  const float* ln_b    = (const float*)d_in[14];
  const float* wq      = (const float*)d_in[15];
  const float* wk      = (const float*)d_in[16];
  const float* wv      = (const float*)d_in[17];
  const float* wo      = (const float*)d_in[18];
  const float* bo      = (const float*)d_in[19];
  const float* ff_w1   = (const float*)d_in[20];
  const float* ff_b1   = (const float*)d_in[21];
  const float* ff_a    = (const float*)d_in[22];
  const float* ff_w2   = (const float*)d_in[23];
  const float* ff_b2   = (const float*)d_in[24];
  const float* dec_w1  = (const float*)d_in[25];
  const float* dec_b1  = (const float*)d_in[26];
  const float* dec_a   = (const float*)d_in[27];
  const float* dec_w2  = (const float*)d_in[28];
  const float* dec_b2  = (const float*)d_in[29];

  float* outp = (float*)d_out;

  // ---- workspace carve (256B aligned) ----
  char* ws = (char*)d_ws;
  size_t off = 0;
  auto alloc = [&](size_t bytes) { size_t o = off; off += (bytes + 255) & ~(size_t)255; return o; };

  float*    h     = (float*)   (ws + alloc((size_t)BT * DMODEL * 4));
  _Float16* hn16  = (_Float16*)(ws + alloc((size_t)BT * DMODEL * 2));
  _Float16* a16   = (_Float16*)(ws + alloc((size_t)BT * DMODEL * 2));
  _Float16* q16   = (_Float16*)(ws + alloc((size_t)BT * DMODEL * 2));
  _Float16* k16   = (_Float16*)(ws + alloc((size_t)BT * DMODEL * 2));
  _Float16* v16t  = (_Float16*)(ws + alloc((size_t)BT * DMODEL * 2));   // [bh][hd][seq]
  _Float16* o16   = (_Float16*)(ws + alloc((size_t)BT * DMODEL * 2));
  _Float16* ff16  = (_Float16*)(ws + alloc((size_t)BT * DFFN * 2));
  _Float16* x16   = (_Float16*)(ws + alloc((size_t)BT * 160 * 2));
  _Float16* rph16 = (_Float16*)(ws + alloc((size_t)1023 * DMODEL * 2));
  _Float16* E16   = (_Float16*)(ws + alloc((size_t)1023 * HD * 2));

  _Float16* enc_w1p = (_Float16*)(ws + alloc((size_t)160 * DMODEL * 2));
  _Float16* enc_w2p = (_Float16*)(ws + alloc((size_t)DMODEL * DMODEL * 2));
  _Float16* rp_w2p  = (_Float16*)(ws + alloc((size_t)DMODEL * HD * 2));
  _Float16* wq16    = (_Float16*)(ws + alloc((size_t)NLAYER * DMODEL * DMODEL * 2));
  _Float16* wk16    = (_Float16*)(ws + alloc((size_t)NLAYER * DMODEL * DMODEL * 2));
  _Float16* wv16    = (_Float16*)(ws + alloc((size_t)NLAYER * DMODEL * DMODEL * 2));
  _Float16* wo16    = (_Float16*)(ws + alloc((size_t)NLAYER * DMODEL * DMODEL * 2));
  _Float16* ffw1p   = (_Float16*)(ws + alloc((size_t)NLAYER * DMODEL * DFFN * 2));
  _Float16* ffw2p   = (_Float16*)(ws + alloc((size_t)NLAYER * DFFN * DMODEL * 2));
  _Float16* dec_w1p = (_Float16*)(ws + alloc((size_t)DMODEL * DMODEL * 2));
  _Float16* dec_w2p = (_Float16*)(ws + alloc((size_t)DMODEL * 80 * 2));

  auto cvtB = [&](const float* src, _Float16* dst, int Kin, int Nin, int Kpad, int Npad) {
    int total = Kpad * Npad;
    cvt_pack_b<<<(total + 255) / 256, 256, 0, stream>>>(src, dst, Kin, Nin, Kpad, Npad);
  };
  auto gemm = [&](const _Float16* A, const _Float16* W, const float* bias,
                  const float* alpha, const float* resid, float* outF,
                  _Float16* outH, _Float16* outHT, float* finalO,
                  int M, int N, int K, int Npack) {
    dim3 grid((N + 63) / 64, (M + 127) / 128);
    gemm_wmma_kernel<<<grid, 128, 0, stream>>>(A, W, bias, alpha, resid,
                                               outF, outH, outHT, finalO, M, N, K, Npack);
  };

  // ---- weight conversion/packing (stacked tensors packed in one shot;
  //      layer stride in packed layout equals the dense per-layer size) ----
  cvtB(enc_w1, enc_w1p, 2 * DOF, DMODEL, 160, DMODEL);
  cvtB(enc_w2, enc_w2p, DMODEL, DMODEL, DMODEL, DMODEL);
  cvtB(rp_w2,  rp_w2p,  DMODEL, HD,     DMODEL, HD);
  cvtB(wq, wq16, NLAYER * DMODEL, DMODEL, NLAYER * DMODEL, DMODEL);
  cvtB(wk, wk16, NLAYER * DMODEL, DMODEL, NLAYER * DMODEL, DMODEL);
  cvtB(wv, wv16, NLAYER * DMODEL, DMODEL, NLAYER * DMODEL, DMODEL);
  cvtB(wo, wo16, NLAYER * DMODEL, DMODEL, NLAYER * DMODEL, DMODEL);
  cvtB(ff_w1, ffw1p, NLAYER * DMODEL, DFFN, NLAYER * DMODEL, DFFN);
  cvtB(ff_w2, ffw2p, NLAYER * DFFN, DMODEL, NLAYER * DFFN, DMODEL);
  cvtB(dec_w1, dec_w1p, DMODEL, DMODEL, DMODEL, DMODEL);
  cvtB(dec_w2, dec_w2p, DMODEL, DOF + 4, DMODEL, 80);

  // ---- encoder ----
  pack_enc<<<(BT * 160 + 255) / 256, 256, 0, stream>>>(x_in, mask_in, x16);
  gemm(x16, enc_w1p, enc_b1, enc_a1, nullptr, nullptr, a16, nullptr, nullptr,
       BT, DMODEL, 160, DMODEL);
  gemm(a16, enc_w2p, enc_b2, enc_a2, nullptr, h, nullptr, nullptr, nullptr,
       BT, DMODEL, DMODEL, DMODEL);

  // ---- relative position table E (1023 x 64) ----
  rp_hidden<<<(1023 * DMODEL + 255) / 256, 256, 0, stream>>>(rp_w1, rp_b1, rp_a, rph16);
  gemm(rph16, rp_w2p, rp_b2, nullptr, nullptr, nullptr, E16, nullptr, nullptr,
       1023, HD, DMODEL, HD);

  // ---- transformer layers ----
  dim3 attGrid(TSEQ / 64, NB * NH);
  for (int l = 0; l < NLAYER; ++l) {
    layernorm_f16<<<BT, 256, 0, stream>>>(h, ln_g, ln_b, hn16);
    const _Float16* Wq = wq16 + (size_t)l * DMODEL * DMODEL;
    const _Float16* Wk = wk16 + (size_t)l * DMODEL * DMODEL;
    const _Float16* Wv = wv16 + (size_t)l * DMODEL * DMODEL;
    const _Float16* Wo = wo16 + (size_t)l * DMODEL * DMODEL;
    gemm(hn16, Wq, nullptr, nullptr, nullptr, nullptr, q16, nullptr, nullptr,
         BT, DMODEL, DMODEL, DMODEL);
    gemm(hn16, Wk, nullptr, nullptr, nullptr, nullptr, k16, nullptr, nullptr,
         BT, DMODEL, DMODEL, DMODEL);
    gemm(hn16, Wv, nullptr, nullptr, nullptr, nullptr, nullptr, v16t, nullptr,
         BT, DMODEL, DMODEL, DMODEL);

    attention_wmma<<<attGrid, 256, ATT_SMEM_BYTES, stream>>>(q16, k16, v16t, E16, o16);

    gemm(o16, Wo, bo + (size_t)l * DMODEL, nullptr, h, h, nullptr, nullptr, nullptr,
         BT, DMODEL, DMODEL, DMODEL);

    layernorm_f16<<<BT, 256, 0, stream>>>(h, ln_g, ln_b, hn16);
    gemm(hn16, ffw1p + (size_t)l * DMODEL * DFFN, ff_b1 + (size_t)l * DFFN,
         ff_a + l, nullptr, nullptr, ff16, nullptr, nullptr, BT, DFFN, DMODEL, DFFN);
    gemm(ff16, ffw2p + (size_t)l * DFFN * DMODEL, ff_b2 + (size_t)l * DMODEL,
         nullptr, h, h, nullptr, nullptr, nullptr, BT, DMODEL, DFFN, DMODEL);
  }

  // ---- decoder ----
  {
    int total = BT * DMODEL;
    cvt_pad_f16<<<(total + 255) / 256, 256, 0, stream>>>(h, a16, BT, DMODEL, BT, DMODEL);
  }
  gemm(a16, dec_w1p, dec_b1, dec_a, nullptr, nullptr, ff16, nullptr, nullptr,
       BT, DMODEL, DMODEL, DMODEL);
  gemm(ff16, dec_w2p, dec_b2, nullptr, nullptr, nullptr, nullptr, nullptr, outp,
       BT, DOF + 4, DMODEL, 80);
}
